// CRFasRNN_Fixed_40355512713231
// MI455X (gfx1250) — compile-verified, matrix operand order checked by reading, not run on hardware
//
#include <hip/hip_runtime.h>
#include <hip/hip_bf16.h>
#include <stdint.h>

typedef __attribute__((ext_vector_type(16))) _Float16 v16h;
typedef __attribute__((ext_vector_type(8)))  float    v8f;

#define Bn  2
#define Cn  21
#define Hn  512
#define Wn  512
#define HWn (Hn*Wn)
#define TS  16      // tile size (pixels per side)
#define Rr  2       // stencil radius
#define TWd 20      // tile + halo
#define CPd 22      // padded channel stride in LDS (dwords) -> conflict-free b64
#define KT  25      // taps

__device__ __forceinline__ int refl(int i, int n) {
  i = (i < 0) ? -i : i;
  return (i >= n) ? (2*n - 2 - i) : i;
}

// CDNA5 async global->LDS copy (ASYNCcnt-tracked), gather-capable per lane.
__device__ __forceinline__ void async_g2l_b32(const void* gsrc, const void* lds_dst) {
  unsigned loff = (unsigned)(unsigned long long)lds_dst; // low 32 bits = LDS offset
  asm volatile("global_load_async_to_lds_b32 %0, %1, off"
               :: "v"(loff), "v"(gsrc) : "memory");
}
__device__ __forceinline__ void wait_async0() {
  asm volatile("s_wait_asynccnt 0" ::: "memory");
}

// ---------------------------------------------------------------------------
// q0 = softmax(unary) over channel axis
// ---------------------------------------------------------------------------
__global__ void k_init_softmax(const float* __restrict__ u, float* __restrict__ q) {
  int p = blockIdx.x * blockDim.x + threadIdx.x;
  if (p >= Bn * HWn) return;
  int b = p / HWn, off = p % HWn;
  const float* ub = u + (size_t)b * Cn * HWn + off;
  float l[Cn];
  float m = -3.4e38f;
#pragma unroll
  for (int c = 0; c < Cn; c++) { l[c] = ub[(size_t)c * HWn]; m = fmaxf(m, l[c]); }
  float s = 0.f;
#pragma unroll
  for (int c = 0; c < Cn; c++) { l[c] = __expf(l[c] - m); s += l[c]; }
  float inv = 1.f / s;
  float* qb = q + (size_t)b * Cn * HWn + off;
#pragma unroll
  for (int c = 0; c < Cn; c++) qb[(size_t)c * HWn] = l[c] * inv;
}

// ---------------------------------------------------------------------------
// wtot[b,k,y,x] = w_rgb[k]/den_rgb + w_edge[k]/den_edge + gauss2d[k]
// (folds both bilateral normalizations and the 5x5 Gaussian into one field)
// ---------------------------------------------------------------------------
__global__ void __launch_bounds__(256)
k_weights(const float* __restrict__ img, const float* __restrict__ edg,
          float* __restrict__ wt) {
  __shared__ float imgt[3][TWd][TWd];
  __shared__ float edgt[TWd][TWd];
  int tx = threadIdx.x, ty = threadIdx.y;
  int tid = ty * TS + tx;
  int x0 = blockIdx.x * TS, y0 = blockIdx.y * TS, b = blockIdx.z;

  // async-stage guide tiles (3ch image + edges) with reflect halo
  for (int i = tid; i < 4 * TWd * TWd; i += TS * TS) {
    int c = i / (TWd * TWd);
    int r = i % (TWd * TWd);
    int yy = r / TWd, xx = r % TWd;
    int gy = refl(y0 + yy - Rr, Hn), gx = refl(x0 + xx - Rr, Wn);
    if (c < 3) async_g2l_b32(&img[(((size_t)b*3 + c)*Hn + gy)*Wn + gx], &imgt[c][yy][xx]);
    else       async_g2l_b32(&edg[((size_t)b*Hn + gy)*Wn + gx],        &edgt[yy][xx]);
  }
  wait_async0();
  __syncthreads();

  // normalized 1-D gaussians: spatial (sigma=5) and blur (sigma=0.5)
  float gs[5], gg[5]; float ss = 0.f, sg = 0.f;
#pragma unroll
  for (int i = 0; i < 5; i++) {
    float xo = (float)(i - 2);
    gs[i] = __expf(-xo * xo * (1.f / 50.f)); ss += gs[i];
    gg[i] = __expf(-xo * xo * 2.f);          sg += gg[i];
  }
  float iss = 1.f / ss, isg = 1.f / sg;
#pragma unroll
  for (int i = 0; i < 5; i++) { gs[i] *= iss; gg[i] *= isg; }

  float c0 = imgt[0][ty+Rr][tx+Rr], c1 = imgt[1][ty+Rr][tx+Rr], c2 = imgt[2][ty+Rr][tx+Rr];
  float ce = edgt[ty+Rr][tx+Rr];
  float w1[KT], w2[KT]; float den1 = 0.f, den2 = 0.f;
#pragma unroll
  for (int dy = 0; dy < 5; dy++)
#pragma unroll
    for (int dx = 0; dx < 5; dx++) {
      int k = dy * 5 + dx;
      float sp = gs[dy] * gs[dx];
      float d = fabsf(imgt[0][ty+dy][tx+dx] - c0)
              + fabsf(imgt[1][ty+dy][tx+dx] - c1)
              + fabsf(imgt[2][ty+dy][tx+dx] - c2);
      w1[k] = sp * __expf(-2.f * d * d);  den1 += w1[k];   // inv2sc2 = -0.5/0.25
      float de = fabsf(edgt[ty+dy][tx+dx] - ce);
      w2[k] = sp * __expf(-2.f * de * de); den2 += w2[k];
    }
  float i1 = 1.f / den1, i2 = 1.f / den2;
  int y = y0 + ty, x = x0 + tx;
  float* wb = wt + (((size_t)b * KT) * Hn + y) * Wn + x;
#pragma unroll
  for (int dy = 0; dy < 5; dy++)
#pragma unroll
    for (int dx = 0; dx < 5; dx++) {
      int k = dy * 5 + dx;
      wb[(size_t)k * HWn] = w1[k] * i1 + w2[k] * i2 + gg[dy] * gg[dx];
    }
}

// ---------------------------------------------------------------------------
// One mean-field iteration, fully fused:
//   s[c]    = sum_k wtot[k,p] * q[c, p+off_k]        (25-tap stencil, LDS tile)
//   colsum  = sum_c s[c]                              (WMMA ones-GEMM, f16-safe)
//   q_out   = softmax_c(unary[c] - (s[c] - colsum))
// ---------------------------------------------------------------------------
__global__ void __launch_bounds__(256)
k_iter(const float* __restrict__ qi, const float* __restrict__ wt,
       const float* __restrict__ u, float* __restrict__ qo) {
  __shared__ __align__(16) float   qt[TWd * TWd * CPd];   // [y][x][c] channel-contig
  __shared__ __align__(64) _Float16 sbuf[TS * TS * 32];   // per-pixel s, padded to 32 ch
  __shared__ float colsum[TS * TS];

  int tx = threadIdx.x, ty = threadIdx.y;
  int tid = ty * TS + tx;
  int x0 = blockIdx.x * TS, y0 = blockIdx.y * TS, b = blockIdx.z;

  // async-stage q tile (21 ch, 20x20 with reflect halo) straight into LDS
  for (int i = tid; i < Cn * TWd * TWd; i += TS * TS) {
    int xx = i % TWd;
    int t  = i / TWd;
    int yy = t % TWd;
    int c  = t / TWd;
    int gy = refl(y0 + yy - Rr, Hn), gx = refl(x0 + xx - Rr, Wn);
    async_g2l_b32(&qi[(((size_t)b * Cn + c) * Hn + gy) * Wn + gx],
                  &qt[(yy * TWd + xx) * CPd + c]);
  }
  wait_async0();
  __syncthreads();

  int y = y0 + ty, x = x0 + tx;
  const float* wb = wt + (((size_t)b * KT) * Hn + y) * Wn + x;
  const float* ub = u  + (((size_t)b * Cn) * Hn + y) * Wn + x;
  __builtin_prefetch(ub, 0, 0);

  float acc[Cn];
#pragma unroll
  for (int c = 0; c < Cn; c++) acc[c] = 0.f;

  // 25-tap per-pixel-weighted stencil; channel-contiguous b64 LDS reads
  for (int k = 0; k < KT; k++) {
    int dy = k / 5, dx = k % 5;
    float wk = wb[(size_t)k * HWn];
    const float* row = &qt[((ty + dy) * TWd + (tx + dx)) * CPd];
#pragma unroll
    for (int cc = 0; cc < 10; cc++) {
      float2 v = *(const float2*)(row + 2 * cc);
      acc[2*cc]     += wk * v.x;
      acc[2*cc + 1] += wk * v.y;
    }
    acc[20] += wk * row[20];
  }

  // channel reduction on the matrix pipe: colsum = ones(1xC) * S(CxP).
  // This term is uniform across channels and cancels exactly in softmax,
  // so f16 inputs do not affect the final output.
#pragma unroll
  for (int c = 0; c < 32; c++)
    sbuf[tid * 32 + c] = (c < Cn) ? (_Float16)acc[c] : (_Float16)0.0f;
  __syncthreads();

  {
    int lane = tid & 31;
    int wv   = tid >> 5;
    v16h a;
#pragma unroll
    for (int j = 0; j < 16; j++) a[j] = (_Float16)1.0f;
#pragma unroll
    for (int i = 0; i < 2; i++) {
      int g  = wv * 2 + i;            // 16-pixel group
      int n  = lane & 15;             // pixel within group (B-matrix column)
      int ks = (lane < 16) ? 0 : 16;  // contiguous K half (k = 16h + s); A is all ones
      v16h bf = *(const v16h*)&sbuf[(g * 16 + n) * 32 + ks];
      v8f d = {};
      d = __builtin_amdgcn_wmma_f32_16x16x32_f16(false, a, false, bf,
                                                 (short)0, d, false, false);
      if (lane < 16) colsum[g * 16 + n] = d[0];  // every D row == channel sum
    }
  }
  __syncthreads();

  // logits = unary - (s - colsum); stable softmax; write q_out
  float cs = colsum[tid];
  float m = -3.4e38f;
#pragma unroll
  for (int c = 0; c < Cn; c++) {
    acc[c] = ub[(size_t)c * HWn] - acc[c] + cs;
    m = fmaxf(m, acc[c]);
  }
  float s = 0.f;
#pragma unroll
  for (int c = 0; c < Cn; c++) { acc[c] = __expf(acc[c] - m); s += acc[c]; }
  float inv = 1.f / s;
  float* qb = qo + (((size_t)b * Cn) * Hn + y) * Wn + x;
#pragma unroll
  for (int c = 0; c < Cn; c++) qb[(size_t)c * HWn] = acc[c] * inv;
}

// ---------------------------------------------------------------------------
extern "C" void kernel_launch(void* const* d_in, const int* in_sizes, int n_in,
                              void* d_out, int out_size, void* d_ws, size_t ws_size,
                              hipStream_t stream) {
  const float* unary = (const float*)d_in[0];
  const float* image = (const float*)d_in[1];
  const float* edges = (const float*)d_in[2];
  float* qout = (float*)d_out;
  float* wtot = (float*)d_ws;                       // 2*25*512*512 f32
  float* qws  = wtot + (size_t)Bn * KT * HWn;       // 2*21*512*512 f32

  dim3 bs(TS, TS, 1);
  dim3 gs(Wn / TS, Hn / TS, Bn);

  k_weights<<<gs, bs, 0, stream>>>(image, edges, wtot);

  int np = Bn * HWn;
  k_init_softmax<<<(np + 255) / 256, 256, 0, stream>>>(unary, qout);

  // ping-pong so iteration 9 (the 10th) lands in d_out
  for (int it = 0; it < 10; ++it) {
    const float* qin = (it & 1) ? qws  : qout;
    float*       qot = (it & 1) ? qout : qws;
    k_iter<<<gs, bs, 0, stream>>>(qin, wtot, unary, qot);
  }
}
